// LSTMModel_64141041598599
// MI455X (gfx1250) — matrix operand order repaired and checked by reading, not compiled, not run on hardware
//
#include <hip/hip_runtime.h>

// ---------------------------------------------------------------------------
// 4-layer LSTM (B=256,T=512,I=64,H=512) + MLP head for MI455X (gfx1250).
//
// ~2 TFLOP of GEMM; HBM traffic is negligible vs 23.3 TB/s -> the binding
// constraint is the 512-step sequential recurrence. Strategy:
//  * bf16 WMMA (v_wmma_f32_16x16x32_bf16): 8x denser than fp32 WMMA, fp32 acc.
//  * one persistent kernel per layer (64 WGs, guaranteed resident) with an
//    atomic grid barrier per time step -> no per-step launch overhead.
//  * cell state c lives in VGPRs for the whole sequence (never in memory).
//  * weights are time-invariant: each WG stages its 4-gate x 16-col weight
//    slice (73.7-128 KB bf16) into LDS once; inner loop B-fragments come from
//    ds_load_b128 (low latency, broadcast to all 8 waves) instead of VMEM.
//  * fused GEMM over concatenated K = [x_t | h_{t-1}] vs packed [W_ih|W_hh],
//    computing all 4 gate pre-activations per wave-tile, cell math per-lane.
// ---------------------------------------------------------------------------

typedef __attribute__((ext_vector_type(16))) __bf16 v16bf;
typedef __attribute__((ext_vector_type(8)))  float  v8f;

constexpr int kB = 256;   // batch
constexpr int kT = 512;   // time steps
constexpr int kI = 64;    // input features
constexpr int kH = 512;   // hidden
constexpr int kG = 2048;  // 4*H gate rows
constexpr int kGridWG = 64;  // 2 (M) x 32 (N) workgroups in the layer kernel

// ----------------------------- helpers -------------------------------------

__device__ __forceinline__ unsigned short f2bf(float f) {
  unsigned int u = __builtin_bit_cast(unsigned int, f);
  unsigned int r = u + 0x7FFFu + ((u >> 16) & 1u);  // round-to-nearest-even
  return (unsigned short)(r >> 16);
}

__device__ __forceinline__ float bf2f(unsigned short s) {
  unsigned int u = ((unsigned int)s) << 16;
  return __builtin_bit_cast(float, u);
}

__device__ __forceinline__ float sigmoid_f(float x) {
  return 1.0f / (1.0f + __expf(-x));
}

__device__ __forceinline__ float tanh_f(float x) {
  return 1.0f - 2.0f / (__expf(2.0f * x) + 1.0f);
}

// A-fragment (16x32 bf16, wave32): lane = (khalf<<4)|m ; elements 0..7 hold
// K = k0+khalf*8+e, elements 8..15 hold K = k0+16+khalf*8+(e-8).
// Caller passes p = row_ptr + k0 + khalf*8.
__device__ __forceinline__ v16bf load_a_frag(const unsigned short* p) {
  union { uint4 q[2]; v16bf v; } u;
  u.q[0] = *(const uint4*)(p);
  u.q[1] = *(const uint4*)(p + 16);
  return u.v;
}

// Monotonic grid barrier (counter zeroed per launch by init kernel).
__device__ __forceinline__ void grid_barrier(unsigned int* ctr, unsigned int target) {
  __syncthreads();
  __threadfence();               // release: make h_t stores visible device-wide
  __syncthreads();
  if (threadIdx.x == 0) {
    atomicAdd(ctr, 1u);
    while (atomicAdd(ctr, 0u) < target) {
      __builtin_amdgcn_s_sleep(1);
    }
  }
  __syncthreads();
  __threadfence();               // acquire: invalidate L0 so h_t reads are fresh
}

// ----------------------------- prep kernels --------------------------------

__global__ void init_ctr_kernel(unsigned int* ctr) {
  if (threadIdx.x < 4) ctr[threadIdx.x] = 0u;
}

// dst[g*(KIN+H)+k] = bf16( k<KIN ? W_ih[g*KIN+k] : W_hh[g*H+(k-KIN)] )
__global__ void pack_weights_kernel(const float* __restrict__ wih,
                                    const float* __restrict__ whh,
                                    unsigned short* __restrict__ dst,
                                    int kin, int n) {
  int tid = blockIdx.x * blockDim.x + threadIdx.x;
  if (tid >= n) return;
  int stride = kin + kH;
  int g = tid / stride;
  int k = tid - g * stride;
  float v = (k < kin) ? wih[g * kin + k] : whh[g * kH + (k - kin)];
  dst[tid] = f2bf(v);
}

__global__ void pack_bias_kernel(const float* __restrict__ bih,
                                 const float* __restrict__ bhh,
                                 float* __restrict__ dst, int n) {
  int tid = blockIdx.x * blockDim.x + threadIdx.x;
  if (tid < n) dst[tid] = bih[tid] + bhh[tid];
}

// x: (B,T,I) f32 -> (T,B,I) bf16
__global__ void convert_x_kernel(const float* __restrict__ x,
                                 unsigned short* __restrict__ dst) {
  int tid = blockIdx.x * blockDim.x + threadIdx.x;  // b*T*I + t*I + i
  int i = tid % kI;
  int t = (tid / kI) % kT;
  int b = tid / (kI * kT);
  dst[(size_t)t * kB * kI + b * kI + i] = f2bf(x[tid]);
}

// ----------------------------- LSTM layer ----------------------------------
//
// Persistent kernel. grid = (32, 2) x 256 threads. Wave w of block (bx,by)
// owns the 16x16 output tile: rows [by*128 + w*16, +16), hidden cols
// [bx*16, +16). Per step: 4 gate accumulators over K = KIN (+ H if t>0),
// then the per-lane cell update; h_t -> out_seq (bf16), c stays in VGPRs.
// B-fragments are served from LDS (weights staged once, reused 512 steps).

template <int KIN>
__global__ __launch_bounds__(256) void lstm_layer_kernel(
    const unsigned short* __restrict__ in_seq,   // (T,B,KIN) bf16
    const unsigned short* __restrict__ wcat,     // (2048, KIN+H) bf16
    const float* __restrict__ bias,              // (2048) f32, b_ih+b_hh
    unsigned short* __restrict__ out_seq,        // (T,B,H) bf16
    unsigned int* __restrict__ ctr) {
  constexpr int kStride = KIN + kH;              // packed weight row length
  constexpr int kWRows  = 64;                    // 4 gates x 16 hidden cols

  extern __shared__ unsigned short lds_w[];      // [kWRows * kStride] bf16

  const int lane  = threadIdx.x & 31;
  const int wave  = threadIdx.x >> 5;
  const int khalf = lane >> 4;
  const int l16   = lane & 15;                    // A-row within tile / D-col

  const int mBase = blockIdx.y * 128 + wave * 16; // batch-row tile base
  const int nBase = blockIdx.x * 16;              // hidden-col tile base

  // ---- stage this WG's weight slice into LDS (once; reused all 512 steps) --
  {
    constexpr int chunksPerRow = kStride / 8;     // uint4 = 8 bf16
    for (int c = threadIdx.x; c < kWRows * chunksPerRow; c += 256) {
      int row = c / chunksPerRow;                 // row = g*16 + n_local
      int ko  = (c - row * chunksPerRow) * 8;
      int g   = row >> 4;
      int nl  = row & 15;
      const unsigned short* src =
          wcat + (size_t)(g * kH + nBase + nl) * kStride + ko;
      *(uint4*)(&lds_w[row * kStride + ko]) = *(const uint4*)src;
    }
  }
  __syncthreads();

  // Per-lane LDS base offsets for the 4 gates' B fragments + bias.
  int wbase[4];
  float bv[4];
#pragma unroll
  for (int g = 0; g < 4; ++g) {
    wbase[g] = (g * 16 + l16) * kStride + khalf * 8;
    bv[g] = bias[g * kH + nBase + l16];
  }

  float cc[8];
#pragma unroll
  for (int r = 0; r < 8; ++r) cc[r] = 0.0f;

  const v8f vzero = {0.f, 0.f, 0.f, 0.f, 0.f, 0.f, 0.f, 0.f};

  for (int t = 0; t < kT; ++t) {
    v8f acc[4];
#pragma unroll
    for (int g = 0; g < 4; ++g) acc[g] = vzero;

    // ---- input segment: x_t (KIN columns) ----
    const unsigned short* xrow =
        in_seq + ((size_t)t * kB + (mBase + l16)) * KIN + khalf * 8;
#pragma unroll 2
    for (int k = 0; k < KIN; k += 32) {
      v16bf afrag = load_a_frag(xrow + k);
#pragma unroll
      for (int g = 0; g < 4; ++g) {
        union { uint4 q[2]; v16bf v; } bu;
        bu.q[0] = *(const uint4*)(&lds_w[wbase[g] + k]);
        bu.q[1] = *(const uint4*)(&lds_w[wbase[g] + k + 16]);
        acc[g] = __builtin_amdgcn_wmma_f32_16x16x32_bf16(
            false, afrag, false, bu.v, (short)0, acc[g], false, false);
      }
    }

    // ---- recurrent segment: h_{t-1} (H columns), zero at t==0 ----
    if (t > 0) {
      const unsigned short* hrow =
          out_seq + ((size_t)(t - 1) * kB + (mBase + l16)) * kH + khalf * 8;
#pragma unroll 2
      for (int k = 0; k < kH; k += 32) {
        v16bf afrag = load_a_frag(hrow + k);
#pragma unroll
        for (int g = 0; g < 4; ++g) {
          union { uint4 q[2]; v16bf v; } bu;
          bu.q[0] = *(const uint4*)(&lds_w[wbase[g] + KIN + k]);
          bu.q[1] = *(const uint4*)(&lds_w[wbase[g] + KIN + k + 16]);
          acc[g] = __builtin_amdgcn_wmma_f32_16x16x32_bf16(
              false, afrag, false, bu.v, (short)0, acc[g], false, false);
        }
      }
    }

    // ---- LSTM cell (per-lane; D layout: vgpr r -> M = khalf*8 + r, N = l16)
    unsigned short* hout = out_seq + (size_t)t * kB * kH;
    const int col = nBase + l16;
#pragma unroll
    for (int r = 0; r < 8; ++r) {
      float gi = sigmoid_f(acc[0][r] + bv[0]);
      float gf = sigmoid_f(acc[1][r] + bv[1]);
      float gg = tanh_f   (acc[2][r] + bv[2]);
      float go = sigmoid_f(acc[3][r] + bv[3]);
      float cp = (t > 0) ? cc[r] : 0.0f;
      float cn = gf * cp + gi * gg;
      cc[r] = cn;
      float hn = go * tanh_f(cn);
      int m = mBase + khalf * 8 + r;
      hout[(size_t)m * kH + col] = f2bf(hn);
    }

    grid_barrier(ctr, (unsigned int)(kGridWG * (t + 1)));
  }
}

// ----------------------------- MLP head ------------------------------------

__global__ __launch_bounds__(128) void head_kernel(
    const unsigned short* __restrict__ hlast,  // (B,H) bf16 (= h[:, T-1, :])
    const float* __restrict__ W1, const float* __restrict__ b1,
    const float* __restrict__ W2, const float* __restrict__ b2,
    const float* __restrict__ W3, const float* __restrict__ b3,
    float* __restrict__ out) {
  __shared__ float s1[128];
  __shared__ float s2[64];
  const int b = blockIdx.x;
  const int j = threadIdx.x;
  const unsigned short* hr = hlast + (size_t)b * kH;

  float a1 = b1[j];
  const float* w1r = W1 + (size_t)j * kH;
  for (int k = 0; k < kH; ++k) a1 += bf2f(hr[k]) * w1r[k];
  s1[j] = fmaxf(a1, 0.0f);
  __syncthreads();

  if (j < 64) {
    float a2 = b2[j];
    const float* w2r = W2 + (size_t)j * 128;
    for (int k = 0; k < 128; ++k) a2 += s1[k] * w2r[k];
    s2[j] = fmaxf(a2, 0.0f);
  }
  __syncthreads();

  if (j == 0) {
    float a3 = b3[0];
    for (int k = 0; k < 64; ++k) a3 += s2[k] * W3[k];
    out[b] = a3;
  }
}

// ----------------------------- launch --------------------------------------

extern "C" void kernel_launch(void* const* d_in, const int* in_sizes, int n_in,
                              void* d_out, int out_size, void* d_ws, size_t ws_size,
                              hipStream_t stream) {
  (void)in_sizes; (void)n_in; (void)out_size; (void)ws_size;

  const float* x        = (const float*)d_in[0];
  const float* W_ih0    = (const float*)d_in[1];
  const float* W_ih_rst = (const float*)d_in[2];
  const float* W_hh     = (const float*)d_in[3];
  const float* b_ih     = (const float*)d_in[4];
  const float* b_hh     = (const float*)d_in[5];
  const float* W1       = (const float*)d_in[6];
  const float* b1       = (const float*)d_in[7];
  const float* W2       = (const float*)d_in[8];
  const float* b2       = (const float*)d_in[9];
  const float* W3       = (const float*)d_in[10];
  const float* b3       = (const float*)d_in[11];
  float* out = (float*)d_out;

  // Workspace carve-up (256B aligned).
  char* ws = (char*)d_ws;
  size_t off = 0;
  auto carve = [&](size_t bytes) -> char* {
    char* p = ws + off;
    off = (off + bytes + 255) & ~(size_t)255;
    return p;
  };
  unsigned short* wcat0 = (unsigned short*)carve((size_t)kG * (kI + kH) * 2);
  unsigned short* wcat1 = (unsigned short*)carve((size_t)kG * (kH + kH) * 2);
  unsigned short* wcat2 = (unsigned short*)carve((size_t)kG * (kH + kH) * 2);
  unsigned short* wcat3 = (unsigned short*)carve((size_t)kG * (kH + kH) * 2);
  float*          bsum  = (float*)carve((size_t)4 * kG * 4);
  unsigned short* xcvt  = (unsigned short*)carve((size_t)kT * kB * kI * 2);
  unsigned short* seqA  = (unsigned short*)carve((size_t)kT * kB * kH * 2);
  unsigned short* seqB  = (unsigned short*)carve((size_t)kT * kB * kH * 2);
  unsigned int*   ctr   = (unsigned int*)carve(4 * sizeof(unsigned int));

  // Prep: reset barrier counters, pack bf16 weights/bias, convert x.
  init_ctr_kernel<<<1, 64, 0, stream>>>(ctr);

  {
    int n = 4 * kG;
    pack_bias_kernel<<<(n + 255) / 256, 256, 0, stream>>>(b_ih, b_hh, bsum, n);
  }
  {
    int n = kB * kT * kI;
    convert_x_kernel<<<(n + 255) / 256, 256, 0, stream>>>(x, xcvt);
  }
  {
    int n0 = kG * (kI + kH);
    pack_weights_kernel<<<(n0 + 255) / 256, 256, 0, stream>>>(
        W_ih0, W_hh + (size_t)0 * kG * kH, wcat0, kI, n0);
    int n1 = kG * (kH + kH);
    pack_weights_kernel<<<(n1 + 255) / 256, 256, 0, stream>>>(
        W_ih_rst + (size_t)0 * kG * kH, W_hh + (size_t)1 * kG * kH, wcat1, kH, n1);
    pack_weights_kernel<<<(n1 + 255) / 256, 256, 0, stream>>>(
        W_ih_rst + (size_t)1 * kG * kH, W_hh + (size_t)2 * kG * kH, wcat2, kH, n1);
    pack_weights_kernel<<<(n1 + 255) / 256, 256, 0, stream>>>(
        W_ih_rst + (size_t)2 * kG * kH, W_hh + (size_t)3 * kG * kH, wcat3, kH, n1);
  }

  // 4 persistent LSTM layers (grid = 32 N-tiles x 2 M-tiles = 64 WGs).
  // Dynamic LDS: 64 weight rows x (KIN+H) bf16 (73.7 KB / 128 KB of 320 KB).
  dim3 grid(kH / 16, kB / 128);
  size_t sh0 = (size_t)64 * (kI + kH) * 2;
  size_t sh1 = (size_t)64 * (kH + kH) * 2;
  lstm_layer_kernel<kI><<<grid, 256, sh0, stream>>>(xcvt, wcat0, bsum + 0 * kG, seqA, ctr + 0);
  lstm_layer_kernel<kH><<<grid, 256, sh1, stream>>>(seqA, wcat1, bsum + 1 * kG, seqB, ctr + 1);
  lstm_layer_kernel<kH><<<grid, 256, sh1, stream>>>(seqB, wcat2, bsum + 2 * kG, seqA, ctr + 2);
  lstm_layer_kernel<kH><<<grid, 256, sh1, stream>>>(seqA, wcat3, bsum + 3 * kG, seqB, ctr + 3);

  // Head on h[:, T-1, :].
  head_kernel<<<kB, 128, 0, stream>>>(seqB + (size_t)(kT - 1) * kB * kH,
                                      W1, b1, W2, b2, W3, b3, out);
}
